// SwinEncoderBlock_43533788513072
// MI455X (gfx1250) — matrix operand order repaired and checked by reading, not compiled, not run on hardware
//
#include <hip/hip_runtime.h>
#include <hip/hip_bf16.h>

// ---------------- types ----------------
typedef __attribute__((ext_vector_type(16))) _Float16 v16h;
typedef __attribute__((ext_vector_type(8)))  float    v8f;
typedef __attribute__((ext_vector_type(8)))  _Float16 h8;
typedef __attribute__((ext_vector_type(4)))  _Float16 h4;
typedef __attribute__((ext_vector_type(4)))  int      v4i;

#define B_   32
#define Himg 56
#define Wimg 56
#define Cdim 512
#define HEADS 16
#define WS 7
#define SS 3
#define HID 2048
#define Ntok 49           // valid tokens per window
#define NPAD 64           // padded tokens per window
#define NW 64             // windows per image
#define HD 32             // head dim
#define NWIN (B_ * NW)    // 2048 total windows
#define MPAD ((size_t)NWIN * NPAD)   // 131072 padded rows
#define MTOK ((size_t)B_ * Himg * Wimg)  // 100352 tokens
#define SCALE 0.17677669529663689f       // HD^-0.5

// gfx1250 async global->LDS path (ASYNCcnt-tracked, VGPR-bypassing DMA)
#if defined(__gfx1250__) && __has_builtin(__builtin_amdgcn_global_load_async_to_lds_b128) && __has_builtin(__builtin_amdgcn_s_wait_asynccnt)
#define USE_ASYNC_LDS 1
#else
#define USE_ASYNC_LDS 0
#endif

// ---------------- helpers ----------------
__device__ __forceinline__ v16h frag16(const _Float16* lo, const _Float16* hi) {
    v16h r;
    *(h8*)&r       = *(const h8*)lo;
    *((h8*)&r + 1) = *(const h8*)hi;
    return r;
}

__device__ __forceinline__ void copy16B_to_lds(const _Float16* g, _Float16* l) {
#if USE_ASYNC_LDS
    __builtin_amdgcn_global_load_async_to_lds_b128(
        (__attribute__((address_space(1))) v4i*)g,
        (__attribute__((address_space(3))) v4i*)l, 0, 0);
#else
    *(h8*)l = *(const h8*)g;
#endif
}

__device__ __forceinline__ void wait_stage() {
#if USE_ASYNC_LDS
    __builtin_amdgcn_s_wait_asynccnt(0);
#endif
    __syncthreads();
}

// ---------------- f32 -> f16 convert ----------------
__global__ void f32_to_f16_k(const float* __restrict__ in, _Float16* __restrict__ out, int n) {
    int i = blockIdx.x * 256 + threadIdx.x;
    if (i < n) out[i] = (_Float16)in[i];
}

// ---------------- LN1 + cyclic shift + window partition -> f16 padded windows ----------------
// one wave per destination (window, token) row; dest h16 is [NWIN, NPAD, C] f16
__global__ __launch_bounds__(256) void ln1_shift_window_k(
    const float* __restrict__ x, const float* __restrict__ g, const float* __restrict__ b,
    _Float16* __restrict__ h16)
{
    int lane = threadIdx.x & 31;
    int wv   = threadIdx.x >> 5;
    size_t dtok = (size_t)blockIdx.x * 8 + wv;          // 0 .. MPAD-1
    int w = (int)(dtok >> 6), n = (int)(dtok & 63);
    _Float16* dst = h16 + dtok * Cdim;
    if (n >= Ntok) {                                    // zero padding rows
        h4 z = {};
        for (int ch = 0; ch < 4; ++ch) *(h4*)(dst + ch * 128 + lane * 4) = z;
        return;
    }
    int bimg = w >> 6, nw = w & 63;
    int i = n / 7, j = n - i * 7;
    int sh = ((nw >> 3) * WS + i + SS) % Himg;
    int sw = ((nw & 7) * WS + j + SS) % Wimg;
    const float* src = x + ((size_t)bimg * (Himg * Wimg) + sh * Wimg + sw) * Cdim;

    float v[16]; float s = 0.f, s2 = 0.f;
    for (int ch = 0; ch < 4; ++ch) {
        float4 f = *(const float4*)(src + ch * 128 + lane * 4);
        v[ch*4+0]=f.x; v[ch*4+1]=f.y; v[ch*4+2]=f.z; v[ch*4+3]=f.w;
        s  += f.x + f.y + f.z + f.w;
        s2 += f.x*f.x + f.y*f.y + f.z*f.z + f.w*f.w;
    }
    for (int off = 1; off < 32; off <<= 1) { s += __shfl_xor(s, off, 32); s2 += __shfl_xor(s2, off, 32); }
    float mean = s * (1.0f / Cdim);
    float var  = s2 * (1.0f / Cdim) - mean * mean;
    float rstd = rsqrtf(var + 1e-5f);
    for (int ch = 0; ch < 4; ++ch) {
        int c = ch * 128 + lane * 4;
        float4 gg = *(const float4*)(g + c);
        float4 bb = *(const float4*)(b + c);
        h4 o;
        o[0] = (_Float16)((v[ch*4+0]-mean)*rstd*gg.x + bb.x);
        o[1] = (_Float16)((v[ch*4+1]-mean)*rstd*gg.y + bb.y);
        o[2] = (_Float16)((v[ch*4+2]-mean)*rstd*gg.z + bb.z);
        o[3] = (_Float16)((v[ch*4+3]-mean)*rstd*gg.w + bb.w);
        *(h4*)(dst + c) = o;
    }
}

// ---------------- LN2: x1 (f32) -> m16 (f16), one wave per token ----------------
__global__ __launch_bounds__(256) void ln2_k(
    const float* __restrict__ x1, const float* __restrict__ g, const float* __restrict__ b,
    _Float16* __restrict__ m16)
{
    int lane = threadIdx.x & 31;
    int wv   = threadIdx.x >> 5;
    size_t t = (size_t)blockIdx.x * 8 + wv;             // 0 .. MTOK-1
    const float* src = x1 + t * Cdim;
    _Float16* dst = m16 + t * Cdim;
    float v[16]; float s = 0.f, s2 = 0.f;
    for (int ch = 0; ch < 4; ++ch) {
        float4 f = *(const float4*)(src + ch * 128 + lane * 4);
        v[ch*4+0]=f.x; v[ch*4+1]=f.y; v[ch*4+2]=f.z; v[ch*4+3]=f.w;
        s  += f.x + f.y + f.z + f.w;
        s2 += f.x*f.x + f.y*f.y + f.z*f.z + f.w*f.w;
    }
    for (int off = 1; off < 32; off <<= 1) { s += __shfl_xor(s, off, 32); s2 += __shfl_xor(s2, off, 32); }
    float mean = s * (1.0f / Cdim);
    float rstd = rsqrtf(s2 * (1.0f / Cdim) - mean * mean + 1e-5f);
    for (int ch = 0; ch < 4; ++ch) {
        int c = ch * 128 + lane * 4;
        float4 gg = *(const float4*)(g + c);
        float4 bb = *(const float4*)(b + c);
        h4 o;
        o[0] = (_Float16)((v[ch*4+0]-mean)*rstd*gg.x + bb.x);
        o[1] = (_Float16)((v[ch*4+1]-mean)*rstd*gg.y + bb.y);
        o[2] = (_Float16)((v[ch*4+2]-mean)*rstd*gg.z + bb.z);
        o[3] = (_Float16)((v[ch*4+3]-mean)*rstd*gg.w + bb.w);
        *(h4*)(dst + c) = o;
    }
}

// ---------------- generic 128x128x32-tiled WMMA GEMM, double-buffered async LDS ----------------
// C[m,n] = sum_k A[m,k] * W[n,k]   (W row-major [N,K], i.e. reference's  h @ W.T)
// MODE 0: qkv   -> +bias, q*=SCALE, scatter to q16/k16 [w,hd,tok,d] and vT16 [w,hd,d,tok]
// MODE 1: proj  -> +bias, window-reverse + reverse-shift, x1 = x + val (f32)
// MODE 2: mlp1  -> +bias, exact GELU, store f16 row-major (ld = N)
// MODE 3: mlp2  -> +bias + x1 residual, store f32 row-major to d_out
template <int MODE>
__global__ __launch_bounds__(256) void gemm_wmma_k(
    const _Float16* __restrict__ A, const _Float16* __restrict__ Wt,
    int M, int N, int K,
    const float* __restrict__ bias,
    const float* __restrict__ f32in, float* __restrict__ f32out,
    _Float16* __restrict__ h0, _Float16* __restrict__ h1, _Float16* __restrict__ h2)
{
    __shared__ _Float16 As[2][128 * 32];
    __shared__ _Float16 Bs[2][128 * 32];

    const int m0 = blockIdx.y * 128;
    const int n0 = blockIdx.x * 128;
    const int tid  = threadIdx.x;
    const int lane = tid & 31;
    const int wv   = tid >> 5;           // 8 waves
    const int wm   = wv & 3;             // 4 waves along M (32 rows each)
    const int wn   = wv >> 2;            // 2 waves along N (64 cols each)
    const int col  = lane & 15;
    const int half = lane >> 4;
    const int kb8  = half * 8;
    const int kb16 = half * 16;

    v8f acc[2][4] = {};

    // stage tile (k-offset k0) into buffer buf: two 16B chunks per thread per matrix
    auto stage = [&](int buf, int k0) {
        for (int c = tid; c < 512; c += 256) {
            int row = c >> 2, kc = (c & 3) << 3;
            copy16B_to_lds(A  + (size_t)(m0 + row) * K + k0 + kc, &As[buf][row * 32 + kc]);
            copy16B_to_lds(Wt + (size_t)(n0 + row) * K + k0 + kc, &Bs[buf][row * 32 + kc]);
        }
    };

    const int nk = K >> 5;
    stage(0, 0);
    wait_stage();

    for (int kt = 0; kt < nk; ++kt) {
        const int cur = kt & 1;
        if (kt + 1 < nk) stage(cur ^ 1, (kt + 1) << 5);   // async prefetch of next tile

        v16h a[2], bfr[4];
        for (int tm = 0; tm < 2; ++tm) {
            const _Float16* p = &As[cur][(wm * 32 + tm * 16 + col) * 32 + kb8];
            a[tm] = frag16(p, p + 16);
        }
        for (int tn = 0; tn < 4; ++tn) {
            const _Float16* p = &Bs[cur][(wn * 64 + tn * 16 + col) * 32 + kb8];
            bfr[tn] = frag16(p, p + 16);
        }
        for (int tm = 0; tm < 2; ++tm)
            for (int tn = 0; tn < 4; ++tn)
                acc[tm][tn] = __builtin_amdgcn_wmma_f32_16x16x32_f16(
                    false, a[tm], false, bfr[tn], (short)0, acc[tm][tn], false, false);

        if (kt + 1 < nk) wait_stage();
    }

    // ------- epilogue -------
    for (int tm = 0; tm < 2; ++tm)
        for (int tn = 0; tn < 4; ++tn)
            for (int r = 0; r < 8; ++r) {
                int gm = m0 + wm * 32 + tm * 16 + half * 8 + r;
                int gn = n0 + wn * 64 + tn * 16 + col;
                float v = acc[tm][tn][r] + bias[gn];
                if (MODE == 0) {
                    int part = gn >> 9, c = gn & 511;
                    int hd = c >> 5, d = c & 31;
                    int w = gm >> 6, tok = gm & 63;
                    if (part == 0)
                        h0[((size_t)((w * 16 + hd) * 64 + tok)) * 32 + d] = (_Float16)(v * SCALE);
                    else if (part == 1)
                        h1[((size_t)((w * 16 + hd) * 64 + tok)) * 32 + d] = (_Float16)v;
                    else
                        h2[((size_t)((w * 16 + hd) * 32 + d)) * 64 + tok] = (_Float16)v;
                } else if (MODE == 1) {
                    int w = gm >> 6, tok = gm & 63;
                    if (tok < Ntok) {
                        int bimg = w >> 6, nw = w & 63;
                        int i = tok / 7, j = tok - i * 7;
                        int dh = ((nw >> 3) * WS + i + SS) % Himg;
                        int dw = ((nw & 7) * WS + j + SS) % Wimg;
                        size_t idx = ((size_t)bimg * (Himg * Wimg) + dh * Wimg + dw) * Cdim + gn;
                        f32out[idx] = f32in[idx] + v;
                    }
                } else if (MODE == 2) {
                    float ge = 0.5f * v * (1.0f + erff(v * 0.70710678118f));
                    h0[(size_t)gm * N + gn] = (_Float16)ge;
                } else {
                    size_t idx = (size_t)gm * N + gn;
                    f32out[idx] = f32in[idx] + v;
                }
            }
}

// ---------------- windowed attention: one block (4 waves) per (window, head) ----------------
// S = q kT (64x64, K=32 == one WMMA K-step), +rel_bias +shift mask, softmax (half-wave
// shfl reductions), P staged via LDS to A-layout, O = P V (K=64, two chained WMMAs)
__global__ __launch_bounds__(128) void attn_k(
    const _Float16* __restrict__ q16, const _Float16* __restrict__ k16,
    const _Float16* __restrict__ vT16, const float* __restrict__ rel_bias,
    _Float16* __restrict__ o16)
{
    __shared__ _Float16 Plds[4 * 16 * 64];   // per-wave 16x64 strip of P

    int w    = blockIdx.x >> 4;
    int head = blockIdx.x & 15;
    int lane = threadIdx.x & 31;
    int wv   = threadIdx.x >> 5;             // row strip [16*wv, 16*wv+16)
    int col  = lane & 15;
    int half = lane >> 4;
    int kb8  = half * 8, kb16 = half * 16;
    int mloc = half * 8;

    const _Float16* qb  = q16  + ((size_t)(w * 16 + head)) * 64 * 32;
    const _Float16* kbp = k16  + ((size_t)(w * 16 + head)) * 64 * 32;
    const _Float16* vb  = vT16 + ((size_t)(w * 16 + head)) * 32 * 64;

    // --- S = q kT ---
    v16h aq;
    { const _Float16* p = qb + (size_t)(wv * 16 + col) * 32 + kb8; aq = frag16(p, p + 16); }
    v8f s[4];
    for (int t = 0; t < 4; ++t) {
        const _Float16* p = kbp + (size_t)(t * 16 + col) * 32 + kb8;
        v16h bk = frag16(p, p + 16);
        v8f z = {};
        s[t] = __builtin_amdgcn_wmma_f32_16x16x32_f16(false, aq, false, bk, (short)0, z, false, false);
    }

    // --- bias + mask + softmax, write P (f16) to LDS ---
    int nw = w & 63, wh = nw >> 3, ww = nw & 7;
    for (int r = 0; r < 8; ++r) {
        int gm = wv * 16 + mloc + r;
        int i1 = gm / 7, j1 = gm - i1 * 7;
        int pr1 = wh * WS + i1, pc1 = ww * WS + j1;
        int l1 = (pr1 < 49 ? 0 : (pr1 < 53 ? 1 : 2)) * 3 + (pc1 < 49 ? 0 : (pc1 < 53 ? 1 : 2));
        float vals[4]; float mx = -3.0e38f;
        for (int t = 0; t < 4; ++t) {
            int gn = t * 16 + col;
            float v = s[t][r];
            if (gm < Ntok && gn < Ntok) {
                int i2 = gn / 7, j2 = gn - i2 * 7;
                int rel = (i1 - i2 + 6) * 13 + (j1 - j2 + 6);
                v += rel_bias[rel * HEADS + head];
                int pr2 = wh * WS + i2, pc2 = ww * WS + j2;
                int l2 = (pr2 < 49 ? 0 : (pr2 < 53 ? 1 : 2)) * 3 + (pc2 < 49 ? 0 : (pc2 < 53 ? 1 : 2));
                if (l1 != l2) v += -100.0f;
            } else {
                v = (gm < Ntok) ? -1.0e30f : 0.0f;   // pad cols -> -inf; pad rows -> harmless uniform
            }
            vals[t] = v;
            mx = fmaxf(mx, v);
        }
        for (int off = 1; off < 16; off <<= 1) mx = fmaxf(mx, __shfl_xor(mx, off, 32));
        float sum = 0.f;
        for (int t = 0; t < 4; ++t) { vals[t] = __expf(vals[t] - mx); sum += vals[t]; }
        for (int off = 1; off < 16; off <<= 1) sum += __shfl_xor(sum, off, 32);
        float inv = 1.0f / sum;
        for (int t = 0; t < 4; ++t)
            Plds[wv * 1024 + (mloc + r) * 64 + t * 16 + col] = (_Float16)(vals[t] * inv);
    }
    __syncthreads();

    // --- O = P V  (K=64 -> two chained WMMAs per 16x16 output tile) ---
    v16h ap[2];
    for (int kt = 0; kt < 2; ++kt) {
        const _Float16* p = Plds + wv * 1024 + col * 64 + kt * 32 + kb8;
        ap[kt] = frag16(p, p + 16);
    }
    v8f o[2] = {};
    for (int nt = 0; nt < 2; ++nt)
        for (int kt = 0; kt < 2; ++kt) {
            const _Float16* p = vb + (size_t)(nt * 16 + col) * 64 + kt * 32 + kb8;
            v16h bv = frag16(p, p + 16);
            o[nt] = __builtin_amdgcn_wmma_f32_16x16x32_f16(false, ap[kt], false, bv, (short)0, o[nt], false, false);
        }

    for (int nt = 0; nt < 2; ++nt)
        for (int r = 0; r < 8; ++r) {
            int gm = wv * 16 + mloc + r;
            int d  = nt * 16 + col;
            o16[((size_t)(w * 64 + gm)) * Cdim + head * 32 + d] = (_Float16)o[nt][r];
        }
}

// ---------------- host: launch sequence ----------------
extern "C" void kernel_launch(void* const* d_in, const int* in_sizes, int n_in,
                              void* d_out, int out_size, void* d_ws, size_t ws_size,
                              hipStream_t stream) {
    const float* x       = (const float*)d_in[0];
    const float* n1g     = (const float*)d_in[1];
    const float* n1b     = (const float*)d_in[2];
    const float* qkv_w   = (const float*)d_in[3];
    const float* qkv_b   = (const float*)d_in[4];
    const float* rel_b   = (const float*)d_in[5];
    const float* proj_w  = (const float*)d_in[6];
    const float* proj_b  = (const float*)d_in[7];
    const float* n2g     = (const float*)d_in[8];
    const float* n2b     = (const float*)d_in[9];
    const float* mlp_w1  = (const float*)d_in[10];
    const float* mlp_b1  = (const float*)d_in[11];
    const float* mlp_w2  = (const float*)d_in[12];
    const float* mlp_b2  = (const float*)d_in[13];
    float* out = (float*)d_out;

    char* ws = (char*)d_ws;
    size_t off = 0;
    auto alloc = [&](size_t bytes) -> void* {
        void* p = ws + off;
        off = (off + bytes + 255) & ~(size_t)255;
        return p;
    };

    _Float16* wqkv16 = (_Float16*)alloc((size_t)1536 * 512 * 2);
    _Float16* wproj16= (_Float16*)alloc((size_t)512 * 512 * 2);
    _Float16* wm1_16 = (_Float16*)alloc((size_t)2048 * 512 * 2);
    _Float16* wm2_16 = (_Float16*)alloc((size_t)512 * 2048 * 2);
    _Float16* h16    = (_Float16*)alloc(MPAD * Cdim * 2);
    _Float16* q16    = (_Float16*)alloc((size_t)NWIN * HEADS * 64 * 32 * 2);
    _Float16* k16    = (_Float16*)alloc((size_t)NWIN * HEADS * 64 * 32 * 2);
    _Float16* vT16   = (_Float16*)alloc((size_t)NWIN * HEADS * 32 * 64 * 2);
    _Float16* o16    = (_Float16*)alloc(MPAD * Cdim * 2);
    float*    x1     = (float*)   alloc(MTOK * Cdim * 4);
    _Float16* m16    = (_Float16*)alloc(MTOK * Cdim * 2);
    _Float16* g16    = (_Float16*)alloc(MTOK * HID * 2);

    // weights f32 -> f16
    { int n = 1536*512; f32_to_f16_k<<<(n+255)/256, 256, 0, stream>>>(qkv_w,  wqkv16,  n); }
    { int n = 512*512;  f32_to_f16_k<<<(n+255)/256, 256, 0, stream>>>(proj_w, wproj16, n); }
    { int n = 2048*512; f32_to_f16_k<<<(n+255)/256, 256, 0, stream>>>(mlp_w1, wm1_16,  n); }
    { int n = 512*2048; f32_to_f16_k<<<(n+255)/256, 256, 0, stream>>>(mlp_w2, wm2_16,  n); }

    // LN1 + shift + window partition (one wave per padded row)
    ln1_shift_window_k<<<(int)(MPAD / 8), 256, 0, stream>>>(x, n1g, n1b, h16);

    // QKV: [131072, 512] x [512, 1536]
    gemm_wmma_k<0><<<dim3(1536/128, (int)(MPAD/128)), 256, 0, stream>>>(
        h16, wqkv16, (int)MPAD, 1536, 512, qkv_b, nullptr, nullptr, q16, k16, vT16);

    // attention: 2048 windows x 16 heads
    attn_k<<<NWIN * HEADS, 128, 0, stream>>>(q16, k16, vT16, rel_b, o16);

    // proj + window-reverse + reverse-shift + residual -> x1
    gemm_wmma_k<1><<<dim3(512/128, (int)(MPAD/128)), 256, 0, stream>>>(
        o16, wproj16, (int)MPAD, 512, 512, proj_b, x, x1, nullptr, nullptr, nullptr);

    // LN2
    ln2_k<<<(int)(MTOK / 8), 256, 0, stream>>>(x1, n2g, n2b, m16);

    // MLP fc1 + GELU
    gemm_wmma_k<2><<<dim3(2048/128, (int)(MTOK/128)), 256, 0, stream>>>(
        m16, wm1_16, (int)MTOK, 2048, 512, mlp_b1, nullptr, nullptr, g16, nullptr, nullptr);

    // MLP fc2 + residual -> out
    gemm_wmma_k<3><<<dim3(512/128, (int)(MTOK/128)), 256, 0, stream>>>(
        g16, wm2_16, (int)MTOK, 512, 2048, mlp_b2, x1, out, nullptr, nullptr, nullptr);
}
